// MultiHeadSelfAttention_86964497809932
// MI455X (gfx1250) — matrix operand order repaired and checked by reading, not compiled, not run on hardware
//
#include <hip/hip_runtime.h>
#include <hip/hip_bf16.h>

// ---------------------------------------------------------------------------
// Multi-head self-attention for MI455X (gfx1250), wave32 + WMMA bf16.
//   B=2, T=2048, D=1024, H=16, Dh=64, fp32 in/out, bf16 WMMA f32-accum inside.
//
// v2: 32x32 register-blocked WMMA GEMMs (2x reuse) and block-cooperative
//     flash attention with double-buffered K/V tiles staged into LDS via
//     global_load_async_to_lds_b128 (ASYNCcnt) — the CDNA5 async copy path.
// ---------------------------------------------------------------------------

typedef __attribute__((ext_vector_type(16))) __bf16 v16bf;
typedef __attribute__((ext_vector_type(8)))  __bf16 v8bf;
typedef __attribute__((ext_vector_type(8)))  float  v8f;

#define WMMA_BF16(a, b, c) \
  __builtin_amdgcn_wmma_f32_16x16x32_bf16(false, (a), false, (b), (short)0, (c), false, false)

#define CAT16(lo, hi) \
  __builtin_shufflevector((lo), (hi), 0,1,2,3,4,5,6,7,8,9,10,11,12,13,14,15)

constexpr int cB  = 2;
constexpr int cT  = 2048;
constexpr int cD  = 1024;
constexpr int cH  = 16;
constexpr int cDh = 64;
constexpr int cHD = cH * cDh;   // 1024

// Async copy of 16 bytes from global memory to LDS (per-lane), tracked by
// ASYNCcnt. LDS destination = low 32 bits of the shared-aperture address
// (ISA 10.2).
__device__ __forceinline__ void async_copy16(void* lds_ptr, const void* gptr) {
  uint32_t loff  = (uint32_t)(uintptr_t)lds_ptr;
  uint64_t gaddr = (uint64_t)(uintptr_t)gptr;
  asm volatile("global_load_async_to_lds_b128 %0, %1, off"
               :: "v"(loff), "v"(gaddr) : "memory");
}
__device__ __forceinline__ void wait_async0() {
  asm volatile("s_wait_asynccnt 0" ::: "memory");
}

// ---------------------------------------------------------------------------
// fp32 -> bf16 elementwise convert
// ---------------------------------------------------------------------------
__global__ void k_cvt(const float* __restrict__ src, __bf16* __restrict__ dst, int n) {
  int i = blockIdx.x * blockDim.x + threadIdx.x;
  if (i < n) dst[i] = (__bf16)src[i];
}

// fp32 [K,N] row-major  ->  bf16 [N,K] row-major (transposed)
__global__ void k_cvt_t(const float* __restrict__ W, __bf16* __restrict__ Wt, int K, int N) {
  int i = blockIdx.x * blockDim.x + threadIdx.x;
  if (i < K * N) {
    int k = i / N, n = i % N;
    Wt[(size_t)n * K + k] = (__bf16)W[i];
  }
}

// ---------------------------------------------------------------------------
// 32x32 register-blocked bf16 WMMA GEMM:  C[M,N] = A[M,K] * Bt[N,K]^T
// One wave computes a 32x32 tile: 2 A-frags x 2 B-frags -> 4 WMMAs per k-step.
//   mode 0: bf16 out, scatter to [B,H,T,Dh]   (Q / K projections)
//   mode 1: bf16 out, scatter to [B,H,Dh,T]   (V projection, transposed)
//   mode 2: fp32 out, plain row-major [M,N]   (output projection)
// ---------------------------------------------------------------------------
__global__ void k_gemm32(const __bf16* __restrict__ A, const __bf16* __restrict__ Bt,
                         void* __restrict__ Out, int M, int N, int K, int mode) {
  const int ln  = threadIdx.x & 31;
  const int wv  = threadIdx.x >> 5;
  const int wid = blockIdx.x * 8 + wv;
  const int tilesM = M >> 5;
  const int tm = wid % tilesM;
  const int tn = wid / tilesM;
  const int m0 = tm << 5;
  const int n0 = tn << 5;
  if (n0 >= N) return;

  const int lane16 = ln & 15;
  const int half   = ln >> 4;        // 0 or 1
  const int lo8    = half << 3;      // unused; A loads also use lo16
  const int lo16   = half << 4;      // B-fragment K-chunk select

  const __bf16* arow0 = A  + (size_t)(m0 + lane16) * K;
  const __bf16* arow1 = A  + (size_t)(m0 + 16 + lane16) * K;
  const __bf16* bcol0 = Bt + (size_t)(n0 + lane16) * K;
  const __bf16* bcol1 = Bt + (size_t)(n0 + 16 + lane16) * K;

  v8f c00 = {}, c01 = {}, c10 = {}, c11 = {};
  for (int kk = 0; kk < K; kk += 32) {
    v16bf a0 = CAT16(*(const v8bf*)(arow0 + kk + lo16),
                     *(const v8bf*)(arow0 + kk + lo16 + 8));
    v16bf a1 = CAT16(*(const v8bf*)(arow1 + kk + lo16),
                     *(const v8bf*)(arow1 + kk + lo16 + 8));
    v16bf b0 = CAT16(*(const v8bf*)(bcol0 + kk + lo16),
                     *(const v8bf*)(bcol0 + kk + lo16 + 8));
    v16bf b1 = CAT16(*(const v8bf*)(bcol1 + kk + lo16),
                     *(const v8bf*)(bcol1 + kk + lo16 + 8));
    c00 = WMMA_BF16(a0, b0, c00);
    c01 = WMMA_BF16(a0, b1, c01);
    c10 = WMMA_BF16(a1, b0, c10);
    c11 = WMMA_BF16(a1, b1, c11);
  }

  const int rbase = half << 3;
  auto store_tile = [&](v8f cc, int mt, int nt) {
    const int col = nt + lane16;
    if (mode == 2) {
      float* O = (float*)Out;
#pragma unroll
      for (int r = 0; r < 8; ++r)
        O[(size_t)(mt + rbase + r) * N + col] = cc[r];
    } else {
      __bf16* O = (__bf16*)Out;
      const int h = col >> 6;     // col / Dh
      const int d = col & 63;     // col % Dh
#pragma unroll
      for (int r = 0; r < 8; ++r) {
        int m = mt + rbase + r;          // global row in [0, B*T)
        int b = m >> 11;                 // m / T
        int t = m & (cT - 1);            // m % T
        size_t idx = (mode == 0)
            ? ((((size_t)b * cH + h) * cT + t) * cDh + d)    // [B,H,T,Dh]
            : ((((size_t)b * cH + h) * cDh + d) * cT + t);   // [B,H,Dh,T]
        O[idx] = (__bf16)cc[r];
      }
    }
  };
  store_tile(c00, m0, n0);
  store_tile(c01, m0, n0 + 16);
  store_tile(c10, m0 + 16, n0);
  store_tile(c11, m0 + 16, n0 + 16);
}

// ---------------------------------------------------------------------------
// Fused flash-style attention, block-cooperative.
//   Block (8 waves) = one (b, h) and 128 queries (16 per wave).
//   Per 32-key block: K-tile (32x64) and V-tile (64x32, transposed) are staged
//   into LDS once per block with global_load_async_to_lds_b128, double-buffered
//   and overlapped with compute (s_wait_asynccnt + barrier handshake).
//   Per wave per key-block:
//     S(16x32) = Q @ K^T                 -> 4 WMMAs (B-frags from LDS)
//     online softmax (shfl_xor row reductions in 16-lane groups)
//     P re-layout C->A via per-wave LDS (s_wait_dscnt for same-wave RAW)
//     O(16x64) += P @ V                  -> 4 WMMAs (B-frags from LDS)
// ---------------------------------------------------------------------------
__global__ void k_attn(const __bf16* __restrict__ Qh, const __bf16* __restrict__ Kh,
                       const __bf16* __restrict__ Vt, const int* __restrict__ amask,
                       __bf16* __restrict__ Obuf) {
  __shared__ __align__(16) __bf16 Kbuf[2][32][64];   // 8 KB: keys x dh
  __shared__ __align__(16) __bf16 Vbuf[2][64][32];   // 8 KB: dh x keys
  __shared__ __align__(16) __bf16 Plds[8][16][32];   // 8 KB: per-wave P staging

  const int tid = threadIdx.x;
  const int ln  = tid & 31;
  const int wv  = tid >> 5;

  const int qb = blockIdx.x & 15;          // 16 query-blocks of 128
  const int h  = (blockIdx.x >> 4) & 15;
  const int b  = blockIdx.x >> 8;
  const size_t bh = (size_t)b * cH + h;
  const int q0 = qb * 128 + wv * 16;       // this wave's 16-query tile

  const int lane16 = ln & 15;
  const int half   = ln >> 4;
  const int lo8    = half << 3;
  const int lo16   = half << 4;
  const float scale = 0.125f;              // Dh^-0.5

  // stage one 32-key K/V tile into LDS buffer `buf` (16 B per thread per tile)
  auto stage = [&](int buf, int k0) {
    {
      int r = tid >> 3;                    // key row 0..31
      int c = (tid & 7) << 3;              // dh chunk {0,8,...,56}
      async_copy16(&Kbuf[buf][r][c], Kh + (bh * cT + k0 + r) * cDh + c);
    }
    {
      int r = tid >> 2;                    // dh row 0..63
      int c = (tid & 3) << 3;              // key chunk {0,8,16,24}
      async_copy16(&Vbuf[buf][r][c], Vt + (bh * cDh + r) * cT + k0 + c);
    }
  };

  // Q A-fragments for K-dim 0..31 and 32..63 (loaded once per wave)
  const __bf16* qrow = Qh + (bh * cT + q0 + lane16) * cDh;
  v16bf aq[2];
#pragma unroll
  for (int s = 0; s < 2; ++s) {
    aq[s] = CAT16(*(const v8bf*)(qrow + s * 32 + lo16),
                  *(const v8bf*)(qrow + s * 32 + lo16 + 8));
  }

  float mrow[8], lrow[8];
#pragma unroll
  for (int r = 0; r < 8; ++r) { mrow[r] = -1.0e30f; lrow[r] = 0.0f; }
  v8f o[4] = {{}, {}, {}, {}};             // O accumulator, 4 d-slices of 16

  stage(0, 0);
  int cur = 0;
  for (int k0 = 0; k0 < cT; k0 += 32) {
    wait_async0();          // this wave's stage ops for buf[cur] complete
    __syncthreads();        // all waves' tiles visible; prev buffer fully read
    if (k0 + 32 < cT) stage(cur ^ 1, k0 + 32);   // overlap next tile with compute

    // ---- S = Q @ K^T (two 16-key halves x two d-steps; B-frags from LDS) ----
    v8f s0 = {}, s1 = {};
#pragma unroll
    for (int ds = 0; ds < 2; ++ds) {
      const __bf16* kr0 = &Kbuf[cur][lane16][ds * 32 + lo16];
      v16bf bk0 = CAT16(*(const v8bf*)kr0, *(const v8bf*)(kr0 + 8));
      s0 = WMMA_BF16(aq[ds], bk0, s0);
      const __bf16* kr1 = &Kbuf[cur][16 + lane16][ds * 32 + lo16];
      v16bf bk1 = CAT16(*(const v8bf*)kr1, *(const v8bf*)(kr1 + 8));
      s1 = WMMA_BF16(aq[ds], bk1, s1);
    }

    // ---- mask addends for this lane's two key columns ----
    const float add0 = (1.0f - (float)amask[b * cT + k0 + lane16])      * -1e9f;
    const float add1 = (1.0f - (float)amask[b * cT + k0 + 16 + lane16]) * -1e9f;

    // ---- online softmax; row stats per-lane for 8 rows (C-layout) ----
#pragma unroll
    for (int r = 0; r < 8; ++r) {
      float v0 = s0[r] * scale + add0;
      float v1 = s1[r] * scale + add1;
      float mx = fmaxf(v0, v1);
#pragma unroll
      for (int msk = 1; msk < 16; msk <<= 1) mx = fmaxf(mx, __shfl_xor(mx, msk, 32));
      float mn   = fmaxf(mrow[r], mx);
      float corr = __expf(mrow[r] - mn);
      float p0   = __expf(v0 - mn);
      float p1   = __expf(v1 - mn);
      float rs   = p0 + p1;
#pragma unroll
      for (int msk = 1; msk < 16; msk <<= 1) rs += __shfl_xor(rs, msk, 32);
      lrow[r] = lrow[r] * corr + rs;
      mrow[r] = mn;
      o[0][r] *= corr; o[1][r] *= corr; o[2][r] *= corr; o[3][r] *= corr;
      int Mr = r + (half << 3);
      Plds[wv][Mr][lane16]      = (__bf16)p0;
      Plds[wv][Mr][16 + lane16] = (__bf16)p1;
    }

    // same-wave LDS RAW: DS pipe is in-order per wave; compiler-barrier + wait
    asm volatile("s_wait_dscnt 0" ::: "memory");

    // ---- P A-fragment (16x32, keys are the K dimension) ----
    const __bf16* prow = &Plds[wv][lane16][0];
    v16bf ap = CAT16(*(const v8bf*)(prow + lo16), *(const v8bf*)(prow + lo16 + 8));

    // ---- O += P @ V (B-frags from LDS V-tile) ----
#pragma unroll
    for (int s = 0; s < 4; ++s) {
      const __bf16* vr = &Vbuf[cur][s * 16 + lane16][lo16];
      v16bf bv = CAT16(*(const v8bf*)vr, *(const v8bf*)(vr + 8));
      o[s] = WMMA_BF16(ap, bv, o[s]);
    }

    cur ^= 1;
  }

  // ---- finalize: O /= l, store bf16 to [B,T,H*Dh] for the out-projection ----
#pragma unroll
  for (int r = 0; r < 8; ++r) {
    float inv = 1.0f / lrow[r];
    int Mr = r + (half << 3);
    size_t orow = ((size_t)b * cT + q0 + Mr) * cHD + (size_t)h * cDh;
#pragma unroll
    for (int s = 0; s < 4; ++s)
      Obuf[orow + s * 16 + lane16] = (__bf16)(o[s][r] * inv);
  }
}

// ---------------------------------------------------------------------------
// Host-side orchestration (all on `stream`, graph-capture safe).
// Workspace layout (48 MB):
//   [ 0, 8M)  xb     bf16 x             [B*T, D]
//   [ 8,10M)  Wqt    bf16 Wq^T          [HD, D]
//   [10,12M)  Wkt    bf16 Wk^T
//   [12,14M)  Wvt    bf16 Wv^T
//   [14,16M)  Wot    bf16 Wo^T          [D, HD]
//   [16,24M)  Qh     bf16 [B,H,T,Dh]
//   [24,32M)  Kh     bf16 [B,H,T,Dh]
//   [32,40M)  Vth    bf16 [B,H,Dh,T]
//   [40,48M)  Obuf   bf16 [B,T,H*Dh]
// ---------------------------------------------------------------------------
extern "C" void kernel_launch(void* const* d_in, const int* in_sizes, int n_in,
                              void* d_out, int out_size, void* d_ws, size_t ws_size,
                              hipStream_t stream) {
  const float* x     = (const float*)d_in[0];
  const int*   amask = (const int*)d_in[1];
  const float* Wq    = (const float*)d_in[2];
  const float* Wk    = (const float*)d_in[3];
  const float* Wv    = (const float*)d_in[4];
  const float* Wo    = (const float*)d_in[5];

  char* ws = (char*)d_ws;
  __bf16* xb   = (__bf16*)(ws);
  __bf16* Wqt  = (__bf16*)(ws + ( 8u << 20));
  __bf16* Wkt  = (__bf16*)(ws + (10u << 20));
  __bf16* Wvt  = (__bf16*)(ws + (12u << 20));
  __bf16* Wot  = (__bf16*)(ws + (14u << 20));
  __bf16* Qh   = (__bf16*)(ws + (16u << 20));
  __bf16* Kh   = (__bf16*)(ws + (24u << 20));
  __bf16* Vth  = (__bf16*)(ws + (32u << 20));
  __bf16* Obuf = (__bf16*)(ws + (40u << 20));

  const int nx = cB * cT * cD;       // 4 Mi elements
  const int nw = cD * cHD;           // 1 Mi elements
  k_cvt  <<<(nx + 255) / 256, 256, 0, stream>>>(x, xb, nx);
  k_cvt_t<<<(nw + 255) / 256, 256, 0, stream>>>(Wq, Wqt, cD, cHD);
  k_cvt_t<<<(nw + 255) / 256, 256, 0, stream>>>(Wk, Wkt, cD, cHD);
  k_cvt_t<<<(nw + 255) / 256, 256, 0, stream>>>(Wv, Wvt, cD, cHD);
  k_cvt_t<<<(nw + 255) / 256, 256, 0, stream>>>(Wo, Wot, cHD, cD);

  const int Mg = cB * cT;                            // 4096 rows
  const int tiles32 = (Mg / 32) * (cHD / 32);        // 4096 waves
  k_gemm32<<<tiles32 / 8, 256, 0, stream>>>(xb, Wqt, Qh,  Mg, cHD, cD, 0);
  k_gemm32<<<tiles32 / 8, 256, 0, stream>>>(xb, Wkt, Kh,  Mg, cHD, cD, 0);
  k_gemm32<<<tiles32 / 8, 256, 0, stream>>>(xb, Wvt, Vth, Mg, cHD, cD, 1);

  const int ablocks = cB * cH * (cT / 128);          // 512 blocks
  k_attn<<<ablocks, 256, 0, stream>>>(Qh, Kh, Vth, amask, Obuf);

  k_gemm32<<<tiles32 / 8, 256, 0, stream>>>(Obuf, Wot, d_out, Mg, cD, cHD, 2);
}
